// IFNet_25211458028124
// MI455X (gfx1250) — compile-verified, matrix operand order checked by reading, not run on hardware
//
#include <hip/hip_runtime.h>

typedef __attribute__((ext_vector_type(16))) _Float16 v16h;
typedef __attribute__((ext_vector_type(8)))  _Float16 v8h;
typedef __attribute__((ext_vector_type(8)))  float    v8f;

// ---------------------------------------------------------------------------
// Zero-fill a small f16 region (zero page for SAME-padding A loads)
// ---------------------------------------------------------------------------
__global__ __launch_bounds__(256)
void k_zero16(_Float16* __restrict__ p, int n)
{
    int i = blockIdx.x * blockDim.x + threadIdx.x;
    if (i < n) p[i] = (_Float16)0.0f;
}

// ---------------------------------------------------------------------------
// Stage: f32 NCHW -> f16 NHWC with channels zero-padded to Cp (multiple of 32)
// ---------------------------------------------------------------------------
__global__ __launch_bounds__(256)
void k_stage(const float* __restrict__ x, _Float16* __restrict__ y,
             int Bn, int C, int H, int W, int Cp)
{
    size_t total  = (size_t)Bn * H * W * Cp;
    size_t stride = (size_t)gridDim.x * blockDim.x;
    for (size_t i = (size_t)blockIdx.x * blockDim.x + threadIdx.x; i < total; i += stride) {
        int c = (int)(i % Cp);
        size_t p = i / Cp;
        int w = (int)(p % W); p /= W;
        int h = (int)(p % H); p /= H;
        int b = (int)p;
        float v = 0.0f;
        if (c < C) v = x[(((size_t)b * C + c) * H + h) * W + w];
        y[i] = (_Float16)v;
    }
}

// ---------------------------------------------------------------------------
// Repack conv weights (OIHW f32) into fragment-major f16 B-matrix tiles:
// layout [ntile][kstep][lane(32)][slot(16)]; K ordering k = tap*Cp + cin,
// kstep = tap*(Cp/32) + chunk.  ntiles here is PADDED to even; co >= Cout
// and cin >= Cin slots are zero-filled so the GEMM loop needs no guards.
// B-fragment (16-bit, 32x16): slot = 2*vgpr + half;
// row k_local = 2*(v&3) + half + (lane>=16)*8 + (v>>2)*16.
// ---------------------------------------------------------------------------
__global__ __launch_bounds__(256)
void k_repack(const float* __restrict__ w, _Float16* __restrict__ frag,
              int Cin, int Cout, int Cp, int ntiles, int ksteps)
{
    size_t total  = (size_t)ntiles * ksteps * 512;
    size_t stride = (size_t)gridDim.x * blockDim.x;
    for (size_t i = (size_t)blockIdx.x * blockDim.x + threadIdx.x; i < total; i += stride) {
        int slot = (int)(i & 15);
        int lane = (int)((i >> 4) & 31);
        size_t r = i >> 9;
        int ks = (int)(r % ksteps);
        int nt = (int)(r / ksteps);
        int n  = lane & 15;
        int hi = lane >> 4;
        int v  = slot >> 1;
        int lo = slot & 1;
        int kl = 2 * (v & 3) + lo + hi * 8 + (v >> 2) * 16;
        int kg = ks * 32 + kl;
        int tap = kg / Cp;
        int cin = kg - tap * Cp;
        int co  = nt * 16 + n;
        float val = 0.0f;
        if (cin < Cin && co < Cout)
            val = w[((size_t)co * Cin + cin) * 9 + tap];
        frag[i] = (_Float16)val;
    }
}

// ---------------------------------------------------------------------------
// conv3x3 (SAME) implicit GEMM via v_wmma_f32_16x16x32_f16.
// One wave32 computes a 16(pixel) x 32(outch) tile: one A fragment feeds two
// independent WMMAs (acc[0]/acc[1]).  Taps fully unrolled (dy/dx constant);
// SAME-padding handled by redirecting OOB lanes' A pointer to a zero page,
// so the inner loop is branch-free: 6 b128 loads + prefetch + 2 WMMAs.
// Epilogue: +bias, optional ReLU, optional post-ReLU residual add.
// ---------------------------------------------------------------------------
template<int CP>
__global__ __launch_bounds__(256)
void k_conv3x3_wmma(const _Float16* __restrict__ xs,   // staged NHWC-Cp f16
                    const _Float16* __restrict__ wf,   // fragment-major weights
                    const _Float16* __restrict__ zpad, // >= 512 zero f16
                    const float* __restrict__ bias,
                    const float* __restrict__ skip,    // NCHW f32 (if hasSkip)
                    float* __restrict__ out,           // NCHW f32
                    int Bn, int H, int W, int Cout, int relu, int hasSkip)
{
    constexpr int CHUNKS = CP / 32;
    constexpr int KSTEPS = 9 * CHUNKS;
    const int P      = H * W;
    const int ptiles = P >> 4;                    // P is always a multiple of 16
    const int ntiles = (Cout + 15) >> 4;
    const int ntp    = (ntiles + 1) & ~1;         // padded (matches repack)
    const int npairs = ntp >> 1;
    const int lane   = threadIdx.x & 31;
    const int wid    = blockIdx.x * (blockDim.x >> 5) + (threadIdx.x >> 5);
    const int total  = Bn * ptiles * npairs;
    if (wid >= total) return;                     // uniform per wave

    int t = wid;
    const int np = t % npairs; t /= npairs;
    const int pt = t % ptiles;
    const int b  = t / ptiles;
    const int nt0 = np << 1;
    const int ln = lane & 15;
    const int hi = lane >> 4;
    const int p0 = pt << 4;

    // A-fragment pixel owned by this lane (row m = lane&15)
    const int pa_pix = p0 + ln;
    const int ah = pa_pix / W;
    const int aw = pa_pix % W;

    const _Float16* __restrict__ w0 =
        wf + (size_t)nt0 * KSTEPS * 512 + (size_t)lane * 16;
    constexpr size_t WSKIP = (size_t)KSTEPS * 512;   // next n-tile
    const _Float16* __restrict__ xpix = xs + (size_t)b * P * CP;
    const _Float16* zp = zpad + hi * 8;

    v8f acc[2] = {};

#pragma unroll
    for (int tap = 0; tap < 9; ++tap) {
        const int dy = tap / 3 - 1;               // constant after unroll
        const int dx = tap % 3 - 1;
        const int hh = ah + dy;
        const int ww = aw + dx;
        const bool inb = ((unsigned)hh < (unsigned)H) & ((unsigned)ww < (unsigned)W);
        // OOB lanes read the zero page: branch-free inner loop, EXEC untouched
        const _Float16* pa =
            inb ? (xpix + ((ptrdiff_t)hh * W + ww) * CP + hi * 8) : zp;

#pragma unroll 4
        for (int c = 0; c < CHUNKS; ++c) {
            union { v16h v; v8h h2[2]; } u;
            u.h2[0] = *(const v8h*)(pa);           // K = kbase..kbase+7   -> slots 0..7
            u.h2[1] = *(const v8h*)(pa + 16);      // K = kbase+16..+23    -> slots 8..15
            pa += 32;

            v16h b0 = *(const v16h*)(w0);          // 32B/lane, 1KB/wave contiguous
            v16h b1 = *(const v16h*)(w0 + WSKIP);
            __builtin_prefetch(w0 + 512, 0, 1);    // -> global_prefetch_b8
            w0 += 512;

            acc[0] = __builtin_amdgcn_wmma_f32_16x16x32_f16(
                         false, u.v, false, b0, (short)0, acc[0], false, false);
            acc[1] = __builtin_amdgcn_wmma_f32_16x16x32_f16(
                         false, u.v, false, b1, (short)0, acc[1], false, false);
        }
    }

    // C/D layout: lane holds channel n = lane&15, rows m = hi*8 + r
#pragma unroll
    for (int q = 0; q < 2; ++q) {
        const int cg = (nt0 + q) * 16 + ln;
        if (cg < Cout) {
            const float bv  = bias[cg];
            size_t obase    = ((size_t)b * Cout + cg) * P + p0 + hi * 8;
            float* po       = out + obase;
            const float* ps = skip + obase;
#pragma unroll
            for (int r = 0; r < 8; ++r) {
                float v = acc[q][r] + bv;
                if (relu)    v = fmaxf(v, 0.0f);
                if (hasSkip) v += ps[r];
                po[r] = v;
            }
        }
    }
}

// ---------------------------------------------------------------------------
// 2x2 average pool, NCHW f32
// ---------------------------------------------------------------------------
__global__ __launch_bounds__(256)
void k_pool2(const float* __restrict__ in, float* __restrict__ out,
             int Bn, int C, int Ho, int Wo)
{
    size_t total  = (size_t)Bn * C * Ho * Wo;
    size_t stride = (size_t)gridDim.x * blockDim.x;
    const int Wi = 2 * Wo;
    for (size_t i = (size_t)blockIdx.x * blockDim.x + threadIdx.x; i < total; i += stride) {
        int w = (int)(i % Wo);
        size_t r = i / Wo;
        int h = (int)(r % Ho);
        size_t bc = r / Ho;
        const float* p = in + (bc * (size_t)(2 * Ho) + 2 * h) * Wi + 2 * w;
        out[i] = 0.25f * (p[0] + p[1] + p[Wi] + p[Wi + 1]);
    }
}

// ---------------------------------------------------------------------------
// Bilinear 2x upsample per reference: coord = o*(H-1)/(2H-1)
// ---------------------------------------------------------------------------
__global__ __launch_bounds__(256)
void k_up2(const float* __restrict__ in, float* __restrict__ out,
           int Bn, int C, int H, int W)
{
    const int Ho = 2 * H, Wo = 2 * W;
    size_t total  = (size_t)Bn * C * Ho * Wo;
    size_t stride = (size_t)gridDim.x * blockDim.x;
    const float sy = (float)(H - 1) / (float)(Ho - 1);
    const float sx = (float)(W - 1) / (float)(Wo - 1);
    for (size_t i = (size_t)blockIdx.x * blockDim.x + threadIdx.x; i < total; i += stride) {
        int ow = (int)(i % Wo);
        size_t r = i / Wo;
        int oh = (int)(r % Ho);
        size_t bc = r / Ho;
        float fy = oh * sy;
        float fx = ow * sx;
        int y0 = (int)fy; float wy = fy - y0; int y1 = (y0 + 1 < H) ? y0 + 1 : H - 1;
        int x0 = (int)fx; float wx = fx - x0; int x1 = (x0 + 1 < W) ? x0 + 1 : W - 1;
        const float* p = in + bc * (size_t)H * W;
        float v00 = p[(size_t)y0 * W + x0], v01 = p[(size_t)y0 * W + x1];
        float v10 = p[(size_t)y1 * W + x0], v11 = p[(size_t)y1 * W + x1];
        out[i] = (v00 * (1.f - wx) + v01 * wx) * (1.f - wy)
               + (v10 * (1.f - wx) + v11 * wx) * wy;
    }
}

// ---------------------------------------------------------------------------
// Final per-pixel separable 51x51 filter on edge-padded frames + channel mean.
// y = sep(i2; k2v,k2h) + sep(i1; k1v,k1h), out = mean_c(y)
// ---------------------------------------------------------------------------
__global__ __launch_bounds__(256)
void k_sepconv(const float* __restrict__ x,    // (Bn,6,H,W): i1=ch0..2, i2=ch3..5
               const float* __restrict__ k2v, const float* __restrict__ k2h,
               const float* __restrict__ k1v, const float* __restrict__ k1h,
               float* __restrict__ out, int Bn, int H, int W)
{
    const int HW = H * W;
    size_t total  = (size_t)Bn * HW;
    size_t stride = (size_t)gridDim.x * blockDim.x;
    for (size_t i = (size_t)blockIdx.x * blockDim.x + threadIdx.x; i < total; i += stride) {
        int w = (int)(i % W);
        int h = (int)((i / W) % H);
        int b = (int)(i / HW);
        float acc = 0.0f;
        for (int pass = 0; pass < 2; ++pass) {
            const float* img = x + ((size_t)b * 6 + (pass == 0 ? 3 : 0)) * HW;
            const float* kv  = (pass == 0 ? k2v : k1v) + (size_t)b * 51 * HW + (size_t)h * W + w;
            const float* kh  = (pass == 0 ? k2h : k1h) + (size_t)b * 51 * HW + (size_t)h * W + w;
            for (int u = 0; u < 51; ++u) {
                int hh = h + u - 25; hh = hh < 0 ? 0 : (hh >= H ? H - 1 : hh);
                const float* r0 = img + (size_t)hh * W;
                const float* r1 = r0 + HW;
                const float* r2 = r1 + HW;
                float kvu = kv[(size_t)u * HW];
                float rowacc = 0.0f;
                for (int v = 0; v < 51; ++v) {
                    int ww = w + v - 25; ww = ww < 0 ? 0 : (ww >= W ? W - 1 : ww);
                    rowacc += kh[(size_t)v * HW] * (r0[ww] + r1[ww] + r2[ww]);
                }
                acc += kvu * rowacc;
            }
        }
        out[i] = acc * (1.0f / 3.0f);
    }
}

// ---------------------------------------------------------------------------
// Host orchestration
// ---------------------------------------------------------------------------
extern "C" void kernel_launch(void* const* d_in, const int* in_sizes, int n_in,
                              void* d_out, int out_size, void* d_ws, size_t ws_size,
                              hipStream_t stream)
{
    (void)in_sizes; (void)out_size; (void)ws_size;
    if (n_in < 95) return;

    // (cin, cout) for the 47 convs in setup_inputs() dict order
    static const int cfg[47][2] = {
        {6,6},{6,6},{6,32},                         // conv32
        {32,32},{32,32},{32,64},                    // conv64
        {64,64},{64,64},{64,128},                   // conv128
        {128,128},{128,128},{128,256},              // conv256
        {256,256},{256,256},{256,512},              // conv512
        {512,512},{512,512},{512,512},              // conv512x512
        {512,512},                                  // upsamp512
        {512,512},{512,512},{512,256},              // upconv256
        {256,256},                                  // upsamp256
        {256,256},{256,256},{256,128},              // upconv128
        {128,128},                                  // upsamp128
        {128,128},{128,128},{128,64},               // upconv64
        {64,64},                                    // upsamp64
        {64,64},{64,64},{64,51},{51,51},            // k2h
        {64,64},{64,64},{64,51},{51,51},            // k2v
        {64,64},{64,64},{64,51},{51,51},            // k1h
        {64,64},{64,64},{64,51},{51,51}             // k1v
    };

    const float* x = (const float*)d_in[0];

    char* base = (char*)d_ws;
    size_t off = 0;
    auto take = [&](size_t bytes) -> void* {
        void* p = base + off;
        off += (bytes + 255) & ~(size_t)255;
        return p;
    };

    // zero page for SAME-padding A loads (1024 f16; CP=512 worst case reads 512)
    _Float16* zpad = (_Float16*)take(2048);
    k_zero16<<<4, 256, 0, stream>>>(zpad, 1024);

    struct Conv {
        const float* w; const float* b;
        int cin, cout, cp, ntp, ksteps;
        _Float16* frag;
    };
    Conv cv[47];
    int ai = 1;
    for (int i = 0; i < 47; ++i) {
        cv[i].cin    = cfg[i][0];
        cv[i].cout   = cfg[i][1];
        cv[i].w      = (const float*)d_in[ai++];
        cv[i].b      = (const float*)d_in[ai++];
        cv[i].cp     = (cv[i].cin + 31) & ~31;
        int ntiles   = (cv[i].cout + 15) >> 4;
        cv[i].ntp    = (ntiles + 1) & ~1;           // pad tiles to even (zero-filled)
        cv[i].ksteps = (9 * cv[i].cp) >> 5;
        size_t n = (size_t)cv[i].ntp * cv[i].ksteps * 512;
        cv[i].frag = (_Float16*)take(n * sizeof(_Float16));
        int blocks = (int)((n + 255) / 256); if (blocks > 4096) blocks = 4096;
        k_repack<<<blocks, 256, 0, stream>>>(cv[i].w, cv[i].frag,
                                             cv[i].cin, cv[i].cout, cv[i].cp,
                                             cv[i].ntp, cv[i].ksteps);
    }

    const int B = 2;
    float*    bufA = (float*)take((size_t)18 << 20);
    float*    bufB = (float*)take((size_t)18 << 20);
    float*    bufC = (float*)take((size_t)28 << 20);
    _Float16* s16  = (_Float16*)take((size_t)18 << 20);
    float* x64  = (float*)take((size_t)B * 64  * 128 * 128 * 4);
    float* x128 = (float*)take((size_t)B * 128 * 64  * 64  * 4);
    float* x256 = (float*)take((size_t)B * 256 * 32  * 32  * 4);
    float* x512 = (float*)take((size_t)B * 512 * 16  * 16  * 4);
    float* kb[4];
    for (int j = 0; j < 4; ++j) kb[j] = (float*)take((size_t)B * 51 * 256 * 256 * 4);

    auto conv = [&](int ci, const float* src, float* dst, int H, int W,
                    int relu, const float* skip) {
        const Conv& c = cv[ci];
        size_t st = (size_t)B * H * W * c.cp;
        int sb = (int)((st + 255) / 256); if (sb > 8192) sb = 8192;
        k_stage<<<sb, 256, 0, stream>>>(src, s16, B, c.cin, H, W, c.cp);
        int waves  = B * ((H * W) >> 4) * (c.ntp >> 1);
        int blocks = (waves + 7) / 8;
        const float* sk = skip ? skip : dst;
        int hs = skip ? 1 : 0;
        switch (c.cp) {
        case 32:
            k_conv3x3_wmma<32><<<blocks, 256, 0, stream>>>(
                s16, c.frag, zpad, c.b, sk, dst, B, H, W, c.cout, relu, hs); break;
        case 64:
            k_conv3x3_wmma<64><<<blocks, 256, 0, stream>>>(
                s16, c.frag, zpad, c.b, sk, dst, B, H, W, c.cout, relu, hs); break;
        case 128:
            k_conv3x3_wmma<128><<<blocks, 256, 0, stream>>>(
                s16, c.frag, zpad, c.b, sk, dst, B, H, W, c.cout, relu, hs); break;
        case 256:
            k_conv3x3_wmma<256><<<blocks, 256, 0, stream>>>(
                s16, c.frag, zpad, c.b, sk, dst, B, H, W, c.cout, relu, hs); break;
        default:
            k_conv3x3_wmma<512><<<blocks, 256, 0, stream>>>(
                s16, c.frag, zpad, c.b, sk, dst, B, H, W, c.cout, relu, hs); break;
        }
    };
    auto pool = [&](const float* src, float* dst, int C, int Ho, int Wo) {
        size_t t = (size_t)B * C * Ho * Wo;
        int blocks = (int)((t + 255) / 256); if (blocks > 8192) blocks = 8192;
        k_pool2<<<blocks, 256, 0, stream>>>(src, dst, B, C, Ho, Wo);
    };
    auto up = [&](const float* src, float* dst, int C, int H, int W) {
        size_t t = (size_t)B * C * 4 * H * W;
        int blocks = (int)((t + 255) / 256); if (blocks > 8192) blocks = 8192;
        k_up2<<<blocks, 256, 0, stream>>>(src, dst, B, C, H, W);
    };

    // ----- encoder -----
    conv(0,  x,    bufA, 256, 256, 1, nullptr);
    conv(1,  bufA, bufB, 256, 256, 1, nullptr);
    conv(2,  bufB, bufA, 256, 256, 1, nullptr);     // 32 @ 256
    pool(bufA, bufB, 32, 128, 128);
    conv(3,  bufB, bufA, 128, 128, 1, nullptr);
    conv(4,  bufA, bufB, 128, 128, 1, nullptr);
    conv(5,  bufB, x64,  128, 128, 1, nullptr);     // skip x64
    pool(x64, bufA, 64, 64, 64);
    conv(6,  bufA, bufB, 64, 64, 1, nullptr);
    conv(7,  bufB, bufA, 64, 64, 1, nullptr);
    conv(8,  bufA, x128, 64, 64, 1, nullptr);       // skip x128
    pool(x128, bufA, 128, 32, 32);
    conv(9,  bufA, bufB, 32, 32, 1, nullptr);
    conv(10, bufB, bufA, 32, 32, 1, nullptr);
    conv(11, bufA, x256, 32, 32, 1, nullptr);       // skip x256
    pool(x256, bufA, 256, 16, 16);
    conv(12, bufA, bufB, 16, 16, 1, nullptr);
    conv(13, bufB, bufA, 16, 16, 1, nullptr);
    conv(14, bufA, x512, 16, 16, 1, nullptr);       // skip x512
    pool(x512, bufA, 512, 8, 8);
    conv(15, bufA, bufB, 8, 8, 1, nullptr);
    conv(16, bufB, bufA, 8, 8, 1, nullptr);
    conv(17, bufA, bufB, 8, 8, 1, nullptr);         // 512 @ 8

    // ----- decoder -----
    up(bufB, bufA, 512, 8, 8);                      // 512 @ 16
    conv(18, bufA, bufB, 16, 16, 1, x512);
    conv(19, bufB, bufA, 16, 16, 1, nullptr);
    conv(20, bufA, bufB, 16, 16, 1, nullptr);
    conv(21, bufB, bufA, 16, 16, 1, nullptr);       // 256 @ 16
    up(bufA, bufB, 256, 16, 16);                    // 256 @ 32
    conv(22, bufB, bufA, 32, 32, 1, x256);
    conv(23, bufA, bufB, 32, 32, 1, nullptr);
    conv(24, bufB, bufA, 32, 32, 1, nullptr);
    conv(25, bufA, bufB, 32, 32, 1, nullptr);       // 128 @ 32
    up(bufB, bufA, 128, 32, 32);                    // 128 @ 64
    conv(26, bufA, bufB, 64, 64, 1, x128);
    conv(27, bufB, bufA, 64, 64, 1, nullptr);
    conv(28, bufA, bufB, 64, 64, 1, nullptr);
    conv(29, bufB, bufA, 64, 64, 1, nullptr);       // 64 @ 64
    up(bufA, bufB, 64, 64, 64);                     // 64 @ 128
    conv(30, bufB, bufA, 128, 128, 1, x64);         // h = bufA (64 @ 128)

    // ----- kernel prediction heads (k2h, k2v, k1h, k1v) -----
    for (int j = 0; j < 4; ++j) {
        int c0 = 31 + j * 4;
        conv(c0 + 0, bufA, bufB, 128, 128, 1, nullptr);
        conv(c0 + 1, bufB, bufC, 128, 128, 1, nullptr);
        conv(c0 + 2, bufC, bufB, 128, 128, 1, nullptr);   // 51 @ 128
        up(bufB, bufC, 51, 128, 128);                     // 51 @ 256
        conv(c0 + 3, bufC, kb[j], 256, 256, 0, nullptr);  // no ReLU
    }

    // ----- separable filtering + channel mean -----
    {
        size_t t = (size_t)B * 256 * 256;
        int blocks = (int)((t + 255) / 256);
        // heads order: kb[0]=k2h kb[1]=k2v kb[2]=k1h kb[3]=k1v
        k_sepconv<<<blocks, 256, 0, stream>>>(x, kb[1], kb[0], kb[3], kb[2],
                                              (float*)d_out, B, 256, 256);
    }
}
